// maxpool_attention_53008486367347
// MI455X (gfx1250) — matrix operand order repaired and checked by reading, not compiled, not run on hardware
//
#include <hip/hip_runtime.h>
#include <hip/hip_bf16.h>

typedef _Float16 v16h __attribute__((ext_vector_type(16)));
typedef _Float16 v8h  __attribute__((ext_vector_type(8)));
typedef float    v8f  __attribute__((ext_vector_type(8)));

#define HWSP   11025      // 105*105
#define P_TOT  88200      // 8*105*105
#define P_PAD  88320      // 690 * 128  (M tiles of 128)
#define CIN    512
#define DD     128
#define BN_EPS 1e-5f
#define CH     5120       // halves per chunk tile (128 rows * 40 stride)
#define CHB    10240      // bytes per chunk tile

// ---------------------------------------------------------------------------
// Weight pre-swizzle into B fragments in this code's contiguous k order (A reads match).
// B (32x16, f16) fragment per wave: lane<16 -> N=lane,    halves j = K 0..15
//                                   lane>=16 -> N=lane-16, halves j = K 16..31
// frag[kc][nt][lane][j], K index kg = tap*512 + ci  (tap = dy*3+dx)
// ---------------------------------------------------------------------------
__global__ __launch_bounds__(256) void prep_wdc_kernel(const float* __restrict__ w,
                                                       _Float16* __restrict__ frag) {
  int idx  = blockIdx.x * 256 + threadIdx.x;   // 144*8*32*16 = 589824 exactly
  int j    = idx & 15;
  int lane = (idx >> 4) & 31;
  int nt   = (idx >> 9) & 7;
  int kc   = idx >> 12;
  int n    = nt * 16 + (lane & 15);
  int kg   = kc * 32 + ((lane < 16) ? 0 : 16) + j;
  int tap  = kg >> 9;
  int ci   = kg & 511;
  int dy   = tap / 3;
  int dx   = tap - dy * 3;
  float v  = w[(((size_t)n * CIN + ci) * 3 + dy) * 3 + dx];
  frag[idx] = (_Float16)v;
}

__global__ __launch_bounds__(256) void prep_wf_kernel(const float* __restrict__ w,
                                                      _Float16* __restrict__ frag) {
  int idx  = blockIdx.x * 256 + threadIdx.x;   // 16*8*32*16 = 65536 exactly
  int j    = idx & 15;
  int lane = (idx >> 4) & 31;
  int nt   = (idx >> 9) & 7;
  int kc   = idx >> 12;
  int n    = nt * 16 + (lane & 15);
  int kg   = kc * 32 + ((lane < 16) ? 0 : 16) + j;
  float v  = w[(size_t)n * CIN + kg];          // wf is (128,512,1,1)
  frag[idx] = (_Float16)v;
}

// ---------------------------------------------------------------------------
// NCHW f32 -> channel-last f16  (xcl[p][c], row = 512 halves = 1 KB)
// ---------------------------------------------------------------------------
__global__ __launch_bounds__(256) void tocl_kernel(const float* __restrict__ x,
                                                   _Float16* __restrict__ xcl) {
  int idx = blockIdx.x * 256 + threadIdx.x;    // cg-major: idx = cg*P_TOT + p
  if (idx >= 64 * P_TOT) return;
  int p  = idx % P_TOT;
  int cg = idx / P_TOT;                        // channel group of 8
  int b  = p / HWSP;
  int q  = p - b * HWSP;
  const float* src = x + ((size_t)b * CIN + (cg << 3)) * HWSP + q;
  v8h d;
#pragma unroll
  for (int j = 0; j < 8; ++j) d[j] = (_Float16)src[(size_t)j * HWSP];
  *(v8h*)(xcl + ((size_t)p << 9) + (cg << 3)) = d;
}

// ---------------------------------------------------------------------------
// conv0: 3x3 conv 512->128, implicit GEMM (M=P, N=128, K=4608) via
// v_wmma_f32_16x16x32_f16.  Workgroup = 8 waves; tile M=128 x N=128;
// each wave owns 4 M-subtiles x 2 N-subtiles (8 accumulators).
// Two K-chunks (64 halves) staged per barrier round with
// GLOBAL_LOAD_ASYNC_TO_LDS_B128 (ASYNCcnt); 16 WMMAs per wave per round.
// Epilogue: +bias, BN, PReLU; writes f16 to cat[p][0..127] (channel-last).
// ---------------------------------------------------------------------------
__global__ __launch_bounds__(256) void conv0_wmma_kernel(
    const _Float16* __restrict__ xcl, const _Float16* __restrict__ wfrag,
    const float* __restrict__ bdc, const float* __restrict__ g0,
    const float* __restrict__ be0, const float* __restrict__ a0,
    _Float16* __restrict__ cat) {
  // Only __shared__ object in this kernel -> LDS offset 0.
  // [buf(2)][chunk(2)][128 rows * 40 halves]
  __shared__ alignas(16) _Float16 atile[4][CH];

  const int tid   = threadIdx.x;
  const int lane  = tid & 31;
  const int wv    = tid >> 5;
  const int mgrp  = wv & 1;             // M half (rows 0-63 / 64-127)
  const int ngrp  = wv >> 1;            // N quarter (2 subtiles)
  const int ptile = blockIdx.x << 7;

  // staging: thread stages 32B (16 halves) of row sm per chunk
  const int sm   = tid >> 1;            // 0..127
  const int soff = (tid & 1) << 4;      // half offset 0 or 16
  const int ps   = ptile + sm;
  const bool pvalid = ps < P_TOT;
  int sb  = ps / HWSP;
  int sq  = ps - sb * HWSP;
  const int syy = sq / 105;
  const int sxx = sq - syy * 105;
  const unsigned ldsBase = (unsigned)((sm * 40 + soff) * 2);  // bytes, buf0/chunk0

  // fragment-read geometry (16-bit A 16x32 layout)
  const int arow = lane & 15;
  const int aoff = (lane < 16) ? 0 : 16;

  // stage K-step t = chunks (2t, 2t+1); both share one conv tap.
  auto stage = [&](int t, int buf) {
    const int tap = t >> 3;                       // 8 steps per tap
    const int dy  = tap / 3 - 1;
    const int dx  = (tap - (tap / 3) * 3) - 1;
    const int iy  = syy + dy, ix = sxx + dx;
    const bool v  = pvalid && ((unsigned)iy < 105u) && ((unsigned)ix < 105u);
    const unsigned lds = ldsBase + (unsigned)buf * (2u * CHB);
    if (v) {
      const _Float16* gp = xcl + ((size_t)(ps + dy * 105 + dx) << 9)
                               + ((t & 7) << 6) + soff;
      asm volatile("global_load_async_to_lds_b128 %0, %1, off"
                   :: "v"(lds), "v"(gp) : "memory");
      asm volatile("global_load_async_to_lds_b128 %0, %1, off"
                   :: "v"(lds + 16u), "v"(gp + 8) : "memory");
      asm volatile("global_load_async_to_lds_b128 %0, %1, off"
                   :: "v"(lds + CHB), "v"(gp + 32) : "memory");
      asm volatile("global_load_async_to_lds_b128 %0, %1, off"
                   :: "v"(lds + CHB + 16u), "v"(gp + 40) : "memory");
    } else {
      *(v8h*)&atile[buf * 2 + 0][sm * 40 + soff]     = v8h{};
      *(v8h*)&atile[buf * 2 + 0][sm * 40 + soff + 8] = v8h{};
      *(v8h*)&atile[buf * 2 + 1][sm * 40 + soff]     = v8h{};
      *(v8h*)&atile[buf * 2 + 1][sm * 40 + soff + 8] = v8h{};
    }
  };

  v8f acc[4][2];
#pragma unroll
  for (int mi = 0; mi < 4; ++mi)
#pragma unroll
    for (int nj = 0; nj < 2; ++nj) acc[mi][nj] = v8f{};

  stage(0, 0);
  for (int t = 0; t < 72; ++t) {
    asm volatile("s_wait_asynccnt 0x0" ::: "memory");
    __syncthreads();
    if (t + 1 < 72) stage(t + 1, (t + 1) & 1);
#pragma unroll
    for (int j = 0; j < 2; ++j) {
      const int kc = 2 * t + j;
      v16h bfr[2];
#pragma unroll
      for (int nj = 0; nj < 2; ++nj) {
        const int nt = ngrp * 2 + nj;
        bfr[nj] = *(const v16h*)(wfrag + (((size_t)(kc * 8 + nt) * 32 + lane) << 4));
      }
      const _Float16* abase = &atile[(t & 1) * 2 + j][0];
#pragma unroll
      for (int mi = 0; mi < 4; ++mi) {
        const _Float16* ap = abase + (mgrp * 64 + mi * 16 + arow) * 40 + aoff;
        v8h lo = *(const v8h*)ap;
        v8h hi = *(const v8h*)(ap + 8);
        v16h a = __builtin_shufflevector(lo, hi, 0, 1, 2, 3, 4, 5, 6, 7,
                                         8, 9, 10, 11, 12, 13, 14, 15);
#pragma unroll
        for (int nj = 0; nj < 2; ++nj) {
          acc[mi][nj] = __builtin_amdgcn_wmma_f32_16x16x32_f16(
              false, a, false, bfr[nj], (short)0, acc[mi][nj], false, false);
        }
      }
    }
  }

  // epilogue: C/D layout -> lane L, vgpr r: M = r + (L>=16?8:0), N = L&15
  const int rbase = (lane < 16) ? 0 : 8;
  const float al  = a0[0];
#pragma unroll
  for (int nj = 0; nj < 2; ++nj) {
    const int n    = ((ngrp * 2 + nj) << 4) + (lane & 15);
    const float bias = bdc[n];
    const float sc   = g0[n] * rsqrtf(1.0f + BN_EPS);
    const float sh   = be0[n];
#pragma unroll
    for (int mi = 0; mi < 4; ++mi) {
#pragma unroll
      for (int r = 0; r < 8; ++r) {
        const int p = ptile + mgrp * 64 + mi * 16 + rbase + r;
        if (p < P_TOT) {
          float y = (acc[mi][nj][r] + bias) * sc + sh;
          y = (y >= 0.0f) ? y : al * y;
          cat[((size_t)p << 9) + n] = (_Float16)y;
        }
      }
    }
  }
}

// ---------------------------------------------------------------------------
// Final 1x1 conv on cat (K=512) -> BN -> PReLU -> f32 NCHW output.
// Same skeleton; A staging is a pure async b128 gather into LDS.
// ---------------------------------------------------------------------------
__global__ __launch_bounds__(256) void final_wmma_kernel(
    const _Float16* __restrict__ cat, const _Float16* __restrict__ wfrag,
    const float* __restrict__ bf, const float* __restrict__ gf,
    const float* __restrict__ bef, const float* __restrict__ af,
    float* __restrict__ out) {
  __shared__ alignas(16) _Float16 atile[4][CH];  // LDS offset 0

  const int tid   = threadIdx.x;
  const int lane  = tid & 31;
  const int wv    = tid >> 5;
  const int mgrp  = wv & 1;
  const int ngrp  = wv >> 1;
  const int ptile = blockIdx.x << 7;

  const int sm   = tid >> 1;            // staged row 0..127
  const int soff = (tid & 1) << 4;      // half offset 0 or 16
  const size_t srow = (size_t)(ptile + sm) << 9;  // rows < P_PAD: in-bounds
  const unsigned ldsBase = (unsigned)((sm * 40 + soff) * 2);

  const int arow = lane & 15;
  const int aoff = (lane < 16) ? 0 : 16;

  auto stage = [&](int t, int buf) {
    const _Float16* gp = cat + srow + (t << 6) + soff;
    const unsigned lds = ldsBase + (unsigned)buf * (2u * CHB);
    asm volatile("global_load_async_to_lds_b128 %0, %1, off"
                 :: "v"(lds), "v"(gp) : "memory");
    asm volatile("global_load_async_to_lds_b128 %0, %1, off"
                 :: "v"(lds + 16u), "v"(gp + 8) : "memory");
    asm volatile("global_load_async_to_lds_b128 %0, %1, off"
                 :: "v"(lds + CHB), "v"(gp + 32) : "memory");
    asm volatile("global_load_async_to_lds_b128 %0, %1, off"
                 :: "v"(lds + CHB + 16u), "v"(gp + 40) : "memory");
  };

  v8f acc[4][2];
#pragma unroll
  for (int mi = 0; mi < 4; ++mi)
#pragma unroll
    for (int nj = 0; nj < 2; ++nj) acc[mi][nj] = v8f{};

  stage(0, 0);
  for (int t = 0; t < 8; ++t) {
    asm volatile("s_wait_asynccnt 0x0" ::: "memory");
    __syncthreads();
    if (t + 1 < 8) stage(t + 1, (t + 1) & 1);
#pragma unroll
    for (int j = 0; j < 2; ++j) {
      const int kc = 2 * t + j;
      v16h bfr[2];
#pragma unroll
      for (int nj = 0; nj < 2; ++nj) {
        const int nt = ngrp * 2 + nj;
        bfr[nj] = *(const v16h*)(wfrag + (((size_t)(kc * 8 + nt) * 32 + lane) << 4));
      }
      const _Float16* abase = &atile[(t & 1) * 2 + j][0];
#pragma unroll
      for (int mi = 0; mi < 4; ++mi) {
        const _Float16* ap = abase + (mgrp * 64 + mi * 16 + arow) * 40 + aoff;
        v8h lo = *(const v8h*)ap;
        v8h hi = *(const v8h*)(ap + 8);
        v16h a = __builtin_shufflevector(lo, hi, 0, 1, 2, 3, 4, 5, 6, 7,
                                         8, 9, 10, 11, 12, 13, 14, 15);
#pragma unroll
        for (int nj = 0; nj < 2; ++nj) {
          acc[mi][nj] = __builtin_amdgcn_wmma_f32_16x16x32_f16(
              false, a, false, bfr[nj], (short)0, acc[mi][nj], false, false);
        }
      }
    }
  }

  const int rbase = (lane < 16) ? 0 : 8;
  const float al  = af[0];
#pragma unroll
  for (int nj = 0; nj < 2; ++nj) {
    const int n    = ((ngrp * 2 + nj) << 4) + (lane & 15);
    const float bias = bf[n];
    const float sc   = gf[n] * rsqrtf(1.0f + BN_EPS);
    const float sh   = bef[n];
#pragma unroll
    for (int mi = 0; mi < 4; ++mi) {
#pragma unroll
      for (int r = 0; r < 8; ++r) {
        const int p = ptile + mgrp * 64 + mi * 16 + rbase + r;
        if (p < P_TOT) {
          float y = (acc[mi][nj][r] + bias) * sc + sh;
          y = (y >= 0.0f) ? y : al * y;
          int b = p / HWSP;
          int q = p - b * HWSP;
          out[((size_t)b * DD + n) * HWSP + q] = y;
        }
      }
    }
  }
}

// ---------------------------------------------------------------------------
// Branch kernels (tiny): pool -> preconv -> qkv -> attn(+BN,PReLU) -> resize
// ---------------------------------------------------------------------------
__global__ void pool_kernel(const float* __restrict__ x, float* __restrict__ pooled,
                            int o, int total) {
  int idx = blockIdx.x * 256 + threadIdx.x;
  if (idx >= total) return;
  int oo = o * o;
  int s  = idx % oo;
  int bc = idx / oo;                    // b*512 + c
  int oy = s / o, ox = s - oy * o;
  int k  = 105 / o;
  const float* base = x + (size_t)bc * HWSP;
  float m = -__builtin_inff();
  for (int yy = oy * k; yy < oy * k + k; ++yy)
    for (int xx = ox * k; xx < ox * k + k; ++xx)
      m = fmaxf(m, base[yy * 105 + xx]);
  pooled[idx] = m;
}

__global__ void preconv_kernel(const float* __restrict__ pooled,
                               const float* __restrict__ wp, const float* __restrict__ bp,
                               float* __restrict__ pre, int o, int total) {
  int idx = blockIdx.x * 256 + threadIdx.x;
  if (idx >= total) return;
  int oo = o * o;
  int s  = idx % oo;
  int bn_ = idx / oo;
  int n  = bn_ % DD;
  int b  = bn_ / DD;
  float acc = bp[n];
  const float* pb = pooled + (size_t)b * CIN * oo + s;
  const float* w  = wp + (size_t)n * CIN;
  for (int c = 0; c < CIN; ++c) acc += pb[(size_t)c * oo] * w[c];
  pre[idx] = acc;                        // [b][n][o*o]
}

__global__ void qkv_kernel(const float* __restrict__ pre,
                           const float* __restrict__ wq, const float* __restrict__ wk,
                           const float* __restrict__ wv,
                           float* __restrict__ qarr, float* __restrict__ karr,
                           float* __restrict__ varr, int o, int total) {
  int idx = blockIdx.x * 256 + threadIdx.x;
  if (idx >= total) return;
  int op  = o + 2;
  int opp = op * op;
  int oo  = o * o;
  int s   = idx % opp;
  int bc  = idx / opp;                   // b*128 + c
  int c   = bc % DD;
  int b   = bc / DD;
  int py  = s / op, px = s - py * op;
  int y   = py - 1, xx = px - 1;
  bool inter = ((unsigned)y < (unsigned)o) && ((unsigned)xx < (unsigned)o);
  float aq = 0.f, ak = 0.f, av = 0.f;
  for (int ci = 0; ci < DD; ++ci) {
    float pv = inter ? pre[((size_t)b * DD + ci) * oo + y * o + xx] : 0.f;
    ak += wk[c * DD + ci] * pv;
    av += wv[c * DD + ci] * pv;
    aq += wq[c * DD + ci] * pv;
  }
  karr[idx] = ak;
  varr[idx] = av;
  if (inter) qarr[(size_t)bc * oo + y * o + xx] = aq;
}

__global__ void attn_kernel(const float* __restrict__ qarr, const float* __restrict__ karr,
                            const float* __restrict__ varr,
                            const float* __restrict__ g, const float* __restrict__ be,
                            const float* __restrict__ a, float* __restrict__ rarr,
                            int o, int total) {
  int idx = blockIdx.x * 256 + threadIdx.x;
  if (idx >= total) return;
  int oo = o * o;
  int op = o + 2;
  int s  = idx % oo;
  int bc = idx / oo;
  int c  = bc % DD;
  int y  = s / o, x_ = s - y * o;
  float q = qarr[idx];
  const float* kb = karr + (size_t)bc * op * op;
  const float* vb = varr + (size_t)bc * op * op;
  float l[9];
  float mx = -__builtin_inff();
#pragma unroll
  for (int t = 0; t < 9; ++t) {
    int di = t / 3, dj = t - di * 3;
    l[t] = q * kb[(y + di) * op + (x_ + dj)];
    mx = fmaxf(mx, l[t]);
  }
  float se = 0.f, ov = 0.f;
#pragma unroll
  for (int t = 0; t < 9; ++t) {
    int di = t / 3, dj = t - di * 3;
    float e = expf(l[t] - mx);
    se += e;
    ov += e * vb[(y + di) * op + (x_ + dj)];
  }
  ov /= se;
  float r = ov * (g[c] * rsqrtf(1.0f + BN_EPS)) + be[c];
  float al = a[0];
  rarr[idx] = (r >= 0.f) ? r : al * r;
}

__global__ void resize_kernel(const float* __restrict__ rarr, _Float16* __restrict__ cat,
                              int o, int choff, int total) {
  int idx = blockIdx.x * 256 + threadIdx.x;
  if (idx >= total) return;
  int s  = idx % HWSP;
  int bc = idx / HWSP;                   // b*128 + c
  int c  = bc % DD;
  int b  = bc / DD;
  int oy = s / 105, ox = s - oy * 105;
  float scale = (float)o / 105.0f;
  float sy = (oy + 0.5f) * scale - 0.5f;
  float sx = (ox + 0.5f) * scale - 0.5f;
  int y0 = (int)floorf(sy);
  int x0 = (int)floorf(sx);
  float wy = sy - (float)y0;
  float wx = sx - (float)x0;
  int y0c = min(max(y0, 0), o - 1), y1c = min(max(y0 + 1, 0), o - 1);
  int x0c = min(max(x0, 0), o - 1), x1c = min(max(x0 + 1, 0), o - 1);
  const float* rb = rarr + (size_t)bc * o * o;
  float v00 = rb[y0c * o + x0c], v01 = rb[y0c * o + x1c];
  float v10 = rb[y1c * o + x0c], v11 = rb[y1c * o + x1c];
  float v0 = v00 + (v01 - v00) * wx;
  float v1 = v10 + (v11 - v10) * wx;
  float val = v0 + (v1 - v0) * wy;
  size_t p = (size_t)b * HWSP + s;
  cat[(p << 9) + choff + c] = (_Float16)val;
}

// ---------------------------------------------------------------------------
static inline int cdiv(int a, int b) { return (a + b - 1) / b; }

extern "C" void kernel_launch(void* const* d_in, const int* in_sizes, int n_in,
                              void* d_out, int out_size, void* d_ws, size_t ws_size,
                              hipStream_t stream) {
  (void)in_sizes; (void)n_in; (void)out_size; (void)ws_size;
  const float* x    = (const float*)d_in[0];
  const float* w_dc = (const float*)d_in[1];
  const float* b_dc = (const float*)d_in[2];
  const float* g0   = (const float*)d_in[3];
  const float* be0  = (const float*)d_in[4];
  const float* a0   = (const float*)d_in[5];
  const float* wf   = (const float*)d_in[30];
  const float* bf   = (const float*)d_in[31];
  const float* gf   = (const float*)d_in[32];
  const float* bef  = (const float*)d_in[33];
  const float* af   = (const float*)d_in[34];
  float* out = (float*)d_out;

  char* ws = (char*)d_ws;
  size_t off = 0;
  auto take = [&](size_t bytes) -> char* {
    char* p = ws + off;
    off = (off + bytes + 255) & ~(size_t)255;
    return p;
  };
  _Float16* wdcFrag = (_Float16*)take((size_t)144 * 8 * 32 * 16 * 2);   // 1.18 MB
  _Float16* wfFrag  = (_Float16*)take((size_t)16 * 8 * 32 * 16 * 2);    // 128 KB
  _Float16* xcl     = (_Float16*)take((size_t)P_TOT * 512 * 2);         // 90.3 MB
  _Float16* cat     = (_Float16*)take((size_t)P_PAD * 512 * 2);         // 90.4 MB
  float* pooled = (float*)take((size_t)8 * 512 * 49 * 4);
  float* pre    = (float*)take((size_t)8 * 128 * 49 * 4);
  float* qarr   = (float*)take((size_t)8 * 128 * 49 * 4);
  float* karr   = (float*)take((size_t)8 * 128 * 81 * 4);
  float* varr   = (float*)take((size_t)8 * 128 * 81 * 4);
  float* rarr   = (float*)take((size_t)8 * 128 * 49 * 4);

  prep_wdc_kernel<<<2304, 256, 0, stream>>>(w_dc, wdcFrag);
  prep_wf_kernel<<<256, 256, 0, stream>>>(wf, wfFrag);
  tocl_kernel<<<cdiv(64 * P_TOT, 256), 256, 0, stream>>>(x, xcl);
  conv0_wmma_kernel<<<690, 256, 0, stream>>>(xcl, wdcFrag, b_dc, g0, be0, a0, cat);

  const int os_[3] = {3, 5, 7};
  for (int bi = 0; bi < 3; ++bi) {
    const int o  = os_[bi];
    const int op = o + 2;
    const float* wp = (const float*)d_in[6 + bi * 8 + 0];
    const float* bp = (const float*)d_in[6 + bi * 8 + 1];
    const float* wq = (const float*)d_in[6 + bi * 8 + 2];
    const float* wk = (const float*)d_in[6 + bi * 8 + 3];
    const float* wv = (const float*)d_in[6 + bi * 8 + 4];
    const float* g  = (const float*)d_in[6 + bi * 8 + 5];
    const float* be = (const float*)d_in[6 + bi * 8 + 6];
    const float* a  = (const float*)d_in[6 + bi * 8 + 7];

    int tpool = 8 * 512 * o * o;
    pool_kernel<<<cdiv(tpool, 256), 256, 0, stream>>>(x, pooled, o, tpool);
    int tpre = 8 * 128 * o * o;
    preconv_kernel<<<cdiv(tpre, 256), 256, 0, stream>>>(pooled, wp, bp, pre, o, tpre);
    int tqkv = 8 * 128 * op * op;
    qkv_kernel<<<cdiv(tqkv, 256), 256, 0, stream>>>(pre, wq, wk, wv, qarr, karr, varr, o, tqkv);
    attn_kernel<<<cdiv(tpre, 256), 256, 0, stream>>>(qarr, karr, varr, g, be, a, rarr, o, tpre);
    int trs = 8 * 128 * HWSP;
    resize_kernel<<<cdiv(trs, 256), 256, 0, stream>>>(rarr, cat, o, 128 * (bi + 1), trs);
  }

  final_wmma_kernel<<<690, 256, 0, stream>>>(cat, wfFrag, bf, gf, bef, af, out);
}
